// MultiHeadMemory_43903155699921
// MI455X (gfx1250) — compile-verified, matrix operand order checked by reading, not run on hardware
//
#include <hip/hip_runtime.h>
#include <hip/hip_bf16.h>

typedef __attribute__((ext_vector_type(16))) __bf16 v16bf;
typedef __attribute__((ext_vector_type(8)))  __bf16 v8bf;
typedef __attribute__((ext_vector_type(8)))  float  v8f;

#define HEADS   8
#define MEMN    65536
#define DIM     128
#define BATCH   256
#define CHUNK   1024
#define NCHUNK  (MEMN / CHUNK)   /* 64 */
#define NSTEP   (CHUNK / 32)     /* 32 */
#define LN_EPS  1e-5f

// ---------------------------------------------------------------------------
// WMMA helpers (CDNA5 wave32, 16x16x32 bf16 -> f32 accum)
// ---------------------------------------------------------------------------
__device__ __forceinline__ v8f wmma_bf16(v16bf a, v16bf b, v8f c) {
  return __builtin_amdgcn_wmma_f32_16x16x32_bf16(false, a, false, b,
                                                 (short)0, c, false, false);
}

// A/B fragment loader from a row-major bf16 row (contiguous along K).
// K-mapping for 16-bit 16x32 fragments: elems 0..7 -> K = k0 + 8*hi + i,
// elems 8..15 -> K = k0 + 16 + 8*hi + i.  (hi = lane>=16)
__device__ __forceinline__ v16bf ldfrag_bf(const __bf16* rowK, int k0, int hi) {
  v8bf lo = *(const v8bf*)(rowK + k0 + hi * 8);
  v8bf hh = *(const v8bf*)(rowK + k0 + 16 + hi * 8);
  v16bf r;
#pragma unroll
  for (int i = 0; i < 8; ++i) { r[i] = lo[i]; r[8 + i] = hh[i]; }
  return r;
}

// Same, but from f32 source with on-the-fly bf16 convert.
__device__ __forceinline__ v16bf ldfrag_f32(const float* rowK, int k0, int hi) {
  v16bf r;
  const float* p0 = rowK + k0 + hi * 8;
  const float* p1 = rowK + k0 + 16 + hi * 8;
#pragma unroll
  for (int i = 0; i < 8; ++i) { r[i] = (__bf16)p0[i]; r[8 + i] = (__bf16)p1[i]; }
  return r;
}

// Streaming (non-temporal) 16-byte bf16 load.
__device__ __forceinline__ v8bf nt_load8(const __bf16* p) {
  return __builtin_nontemporal_load((v8bf*)p);
}

// Butterfly reductions across a 16-lane half-wave (matches C/D layout halves).
__device__ __forceinline__ float hsum16(float x) {
#pragma unroll
  for (int m = 1; m < 16; m <<= 1) x += __shfl_xor(x, m, 16);
  return x;
}
__device__ __forceinline__ float hmax16(float x) {
#pragma unroll
  for (int m = 1; m < 16; m <<= 1) x = fmaxf(x, __shfl_xor(x, m, 16));
  return x;
}

// ---------------------------------------------------------------------------
// Kernel 1: k = softmax(LN(mem @ fk_w^T + fk_b)); v = relu(LN(mem @ fv_w^T + fv_b))
// Output bf16.  Block = 256 thr (8 waves), each wave does 4 row-tiles of 16.
// Grid = 8*65536/16 / 32 = 1024 blocks.
// ---------------------------------------------------------------------------
__global__ __launch_bounds__(256) void kv_kernel(
    const float* __restrict__ mem,
    const float* __restrict__ fk_w, const float* __restrict__ fk_b,
    const float* __restrict__ fv_w, const float* __restrict__ fv_b,
    const float* __restrict__ nk_g, const float* __restrict__ nk_b,
    const float* __restrict__ nv_g, const float* __restrict__ nv_b,
    __bf16* __restrict__ k_bf, __bf16* __restrict__ v_bf)
{
  __shared__ __bf16 wk[DIM * DIM];
  __shared__ __bf16 wv[DIM * DIM];
  const int tid = threadIdx.x;
  for (int i = tid; i < DIM * DIM; i += 256) {
    wk[i] = (__bf16)fk_w[i];
    wv[i] = (__bf16)fv_w[i];
  }
  __syncthreads();

  const int wave = tid >> 5;
  const int l    = tid & 31;
  const int hi   = l >> 4;
  const int ln   = l & 15;

  for (int w = 0; w < 4; ++w) {
    const long long tile = (long long)(blockIdx.x * 8 + wave) * 4 + w;
    const long long g    = tile * 16;              // flat row: h*65536 + n
    const float* rowp    = mem + (g + ln) * DIM;

    v16bf af[4];
#pragma unroll
    for (int ks = 0; ks < 4; ++ks) af[ks] = ldfrag_f32(rowp, ks * 32, hi);

#pragma unroll
    for (int path = 0; path < 2; ++path) {
      const __bf16* wgt = path == 0 ? wk : wv;
      const float*  bia = path == 0 ? fk_b : fv_b;
      const float*  lng = path == 0 ? nk_g : nv_g;
      const float*  lnb = path == 0 ? nk_b : nv_b;
      __bf16* dst       = path == 0 ? k_bf : v_bf;

      v8f acc[8];
#pragma unroll
      for (int nt = 0; nt < 8; ++nt) {
        v8f a = {};
#pragma unroll
        for (int ks = 0; ks < 4; ++ks) {
          v16bf b = ldfrag_bf(&wgt[(nt * 16 + ln) * DIM], ks * 32, hi);
          a = wmma_bf16(af[ks], b, a);
        }
        acc[nt] = a;
      }
      // bias
#pragma unroll
      for (int nt = 0; nt < 8; ++nt) {
        float bb = bia[nt * 16 + ln];
#pragma unroll
        for (int r = 0; r < 8; ++r) acc[nt][r] += bb;
      }
      // LayerNorm across 128 cols of each row (8 nt * 16 lanes)
      float s1[8], s2[8];
#pragma unroll
      for (int r = 0; r < 8; ++r) { s1[r] = 0.f; s2[r] = 0.f; }
#pragma unroll
      for (int nt = 0; nt < 8; ++nt)
#pragma unroll
        for (int r = 0; r < 8; ++r) { float x = acc[nt][r]; s1[r] += x; s2[r] += x * x; }
      float mu[8], rs[8];
#pragma unroll
      for (int r = 0; r < 8; ++r) {
        s1[r] = hsum16(s1[r]); s2[r] = hsum16(s2[r]);
        mu[r] = s1[r] * (1.f / 128.f);
        float var = s2[r] * (1.f / 128.f) - mu[r] * mu[r];
        rs[r] = rsqrtf(var + LN_EPS);
      }
#pragma unroll
      for (int nt = 0; nt < 8; ++nt) {
        float gg = lng[nt * 16 + ln], bb = lnb[nt * 16 + ln];
#pragma unroll
        for (int r = 0; r < 8; ++r)
          acc[nt][r] = (acc[nt][r] - mu[r]) * rs[r] * gg + bb;
      }

      if (path == 0) {
        // softmax over the 128 key-dim columns
        float mx[8], sm[8];
#pragma unroll
        for (int r = 0; r < 8; ++r) mx[r] = -3.0e38f;
#pragma unroll
        for (int nt = 0; nt < 8; ++nt)
#pragma unroll
          for (int r = 0; r < 8; ++r) mx[r] = fmaxf(mx[r], acc[nt][r]);
#pragma unroll
        for (int r = 0; r < 8; ++r) { mx[r] = hmax16(mx[r]); sm[r] = 0.f; }
#pragma unroll
        for (int nt = 0; nt < 8; ++nt)
#pragma unroll
          for (int r = 0; r < 8; ++r) {
            float e = __expf(acc[nt][r] - mx[r]); acc[nt][r] = e; sm[r] += e;
          }
#pragma unroll
        for (int r = 0; r < 8; ++r) sm[r] = 1.f / hsum16(sm[r]);
#pragma unroll
        for (int nt = 0; nt < 8; ++nt)
#pragma unroll
          for (int r = 0; r < 8; ++r) acc[nt][r] *= sm[r];
      } else {
        // relu
#pragma unroll
        for (int nt = 0; nt < 8; ++nt)
#pragma unroll
          for (int r = 0; r < 8; ++r) acc[nt][r] = fmaxf(acc[nt][r], 0.f);
      }

      // store bf16 (streaming), D-layout: row = g + r + 8*hi, col = nt*16 + ln
#pragma unroll
      for (int nt = 0; nt < 8; ++nt)
#pragma unroll
        for (int r = 0; r < 8; ++r)
          __builtin_nontemporal_store(
              (__bf16)acc[nt][r],
              &dst[(g + r + 8 * hi) * DIM + nt * 16 + ln]);
    }
  }
}

// ---------------------------------------------------------------------------
// Kernel 2: flash attention, split over n.  Grid (NCHUNK, HEADS), 512 thr.
// Each of 16 waves owns 16 batch rows; block streams a 1024-slot chunk with
// register double-buffering of the k/v tiles to hide global-load latency.
// ---------------------------------------------------------------------------
__global__ __launch_bounds__(512) void attn_kernel(
    const float* __restrict__ q,
    const __bf16* __restrict__ k_bf, const __bf16* __restrict__ v_bf,
    float* __restrict__ o_part, float* __restrict__ m_part,
    float* __restrict__ l_part)
{
  __shared__ __bf16 k_lds[32][DIM];       // 8 KB, row-major [n][kidx]
  __shared__ __bf16 vT[DIM][40];          // 10 KB, transposed [vcol][n], padded
  __shared__ __bf16 p_lds[16][16][32];    // 16 KB, per-wave P transpose buffer

  const int c    = blockIdx.x;
  const int h    = blockIdx.y;
  const int tid  = threadIdx.x;
  const int wave = tid >> 5;
  const int l    = tid & 31;
  const int hi   = l >> 4;
  const int ln   = l & 15;

  // q A-fragments for this wave's 16 batch rows (f32 -> bf16)
  const float* qrow = q + (long long)(wave * 16 + ln) * DIM;
  v16bf aq[4];
#pragma unroll
  for (int ks = 0; ks < 4; ++ks) aq[ks] = ldfrag_f32(qrow, ks * 32, hi);

  float m[8], ls[8];
  v8f oacc[8];
  v8f vz = {};
#pragma unroll
  for (int r = 0; r < 8; ++r) { m[r] = -3.0e38f; ls[r] = 0.f; }
#pragma unroll
  for (int vt = 0; vt < 8; ++vt) oacc[vt] = vz;

  const long long kvbase = ((long long)h * MEMN + (long long)c * CHUNK) * DIM;

  // this thread's staging slice: 16 bytes of k and of v per step
  const int sn  = tid >> 4;          // row 0..31 within tile
  const int se8 = (tid & 15) * 8;    // column start 0..120

  // prologue: prefetch tile 0 into registers (streaming loads)
  v8bf kreg = nt_load8(k_bf + kvbase + (long long)sn * DIM + se8);
  v8bf vreg = nt_load8(v_bf + kvbase + (long long)sn * DIM + se8);

  for (int s = 0; s < NSTEP; ++s) {
    __syncthreads();  // previous tile fully consumed by all waves
    // commit prefetched tile to LDS
    *(v8bf*)&k_lds[sn][se8] = kreg;
#pragma unroll
    for (int j = 0; j < 8; ++j) vT[se8 + j][sn] = vreg[j];
    // issue next tile's global loads now; latency hides behind compute(s)
    if (s + 1 < NSTEP) {
      const long long off =
          kvbase + (long long)((s + 1) * 32 + sn) * DIM + se8;
      kreg = nt_load8(k_bf + off);
      vreg = nt_load8(v_bf + off);
    }
    __syncthreads();

    // GEMM1: a[16 batch x 32 slots] = q . k^T
    v8f a0 = vz, a1 = vz;
#pragma unroll
    for (int ks = 0; ks < 4; ++ks) {
      v16bf b0 = ldfrag_bf(&k_lds[ln][0],      ks * 32, hi);
      v16bf b1 = ldfrag_bf(&k_lds[16 + ln][0], ks * 32, hi);
      a0 = wmma_bf16(aq[ks], b0, a0);
      a1 = wmma_bf16(aq[ks], b1, a1);
    }

    // online softmax update per row
    float p0[8], p1[8];
#pragma unroll
    for (int r = 0; r < 8; ++r) {
      float t  = hmax16(fmaxf(a0[r], a1[r]));
      float mn = fmaxf(m[r], t);
      float sc = __expf(m[r] - mn);
      p0[r] = __expf(a0[r] - mn);
      p1[r] = __expf(a1[r] - mn);
      float u = hsum16(p0[r] + p1[r]);
      ls[r] = ls[r] * sc + u;
      m[r]  = mn;
#pragma unroll
      for (int vt = 0; vt < 8; ++vt) oacc[vt][r] *= sc;
    }

    // transpose P from D-layout to A-layout via per-wave LDS
#pragma unroll
    for (int r = 0; r < 8; ++r) {
      p_lds[wave][r + 8 * hi][ln]      = (__bf16)p0[r];
      p_lds[wave][r + 8 * hi][16 + ln] = (__bf16)p1[r];
    }
    asm volatile("s_wait_dscnt 0" ::: "memory");
    v16bf pf = ldfrag_bf(&p_lds[wave][ln][0], 0, hi);

    // GEMM2: oacc += P . v
#pragma unroll
    for (int vt = 0; vt < 8; ++vt) {
      v16bf bv = ldfrag_bf(&vT[vt * 16 + ln][0], 0, hi);
      oacc[vt] = wmma_bf16(pf, bv, oacc[vt]);
    }
  }

  // write per-chunk partials (streaming stores)
  const long long pbase =
      ((long long)(h * NCHUNK + c) * BATCH + wave * 16) * DIM;
#pragma unroll
  for (int vt = 0; vt < 8; ++vt)
#pragma unroll
    for (int r = 0; r < 8; ++r)
      __builtin_nontemporal_store(
          oacc[vt][r],
          &o_part[pbase + (long long)(r + 8 * hi) * DIM + vt * 16 + ln]);
  if (ln < 8) {
    const int r = ln;
    const long long sidx =
        (long long)(h * NCHUNK + c) * BATCH + wave * 16 + r + 8 * hi;
    m_part[sidx] = m[r];
    l_part[sidx] = ls[r];
  }
}

// ---------------------------------------------------------------------------
// Kernel 3: recombine the 64 chunks per (h,b): log-sum-exp merge.
// Grid = HEADS*BATCH blocks of 128 threads (one per v-column).
// ---------------------------------------------------------------------------
__global__ __launch_bounds__(128) void reduce_kernel(
    const float* __restrict__ o_part, const float* __restrict__ m_part,
    const float* __restrict__ l_part, __bf16* __restrict__ x_bf)
{
  __shared__ float sm[NCHUNK], sl[NCHUNK];
  const int b = blockIdx.x & (BATCH - 1);
  const int h = blockIdx.x >> 8;
  const int t = threadIdx.x;
  if (t < NCHUNK) {
    const long long sidx = (long long)(h * NCHUNK + t) * BATCH + b;
    sm[t] = m_part[sidx];
    sl[t] = l_part[sidx];
  }
  __syncthreads();
  float M = -3.0e38f;
  for (int cc = 0; cc < NCHUNK; ++cc) M = fmaxf(M, sm[cc]);
  float L = 0.f;
  for (int cc = 0; cc < NCHUNK; ++cc) L += sl[cc] * __expf(sm[cc] - M);
  float O = 0.f;
  for (int cc = 0; cc < NCHUNK; ++cc) {
    const float w = __expf(sm[cc] - M);
    O += __builtin_nontemporal_load(
             &o_part[((long long)(h * NCHUNK + cc) * BATCH + b) * DIM + t]) * w;
  }
  x_bf[(long long)b * (HEADS * DIM) + h * DIM + t] = (__bf16)(O / L);
}

// ---------------------------------------------------------------------------
// Kernel 4: out = relu(LN(x @ fx_w^T + fx_b)).  16 blocks x 1 wave,
// each wave does a 16-row tile of the 256x128 output via bf16 WMMA (K=1024).
// ---------------------------------------------------------------------------
__global__ __launch_bounds__(32) void proj_kernel(
    const __bf16* __restrict__ x_bf,
    const float* __restrict__ fx_w, const float* __restrict__ fx_b,
    const float* __restrict__ nx_g, const float* __restrict__ nx_b,
    float* __restrict__ out)
{
  const int l  = threadIdx.x & 31;
  const int hi = l >> 4;
  const int ln = l & 15;
  const int base = blockIdx.x * 16;

  v8f acc[8];
  v8f vz = {};
#pragma unroll
  for (int nt = 0; nt < 8; ++nt) acc[nt] = vz;

  const __bf16* xrow = x_bf + (long long)(base + ln) * (HEADS * DIM);
  for (int ks = 0; ks < 32; ++ks) {
    v16bf a = ldfrag_bf(xrow, ks * 32, hi);
#pragma unroll
    for (int nt = 0; nt < 8; ++nt) {
      v16bf b = ldfrag_f32(fx_w + (long long)(nt * 16 + ln) * (HEADS * DIM),
                           ks * 32, hi);
      acc[nt] = wmma_bf16(a, b, acc[nt]);
    }
  }
  // bias
#pragma unroll
  for (int nt = 0; nt < 8; ++nt) {
    float bb = fx_b[nt * 16 + ln];
#pragma unroll
    for (int r = 0; r < 8; ++r) acc[nt][r] += bb;
  }
  // LayerNorm + relu
  float s1[8], s2[8];
#pragma unroll
  for (int r = 0; r < 8; ++r) { s1[r] = 0.f; s2[r] = 0.f; }
#pragma unroll
  for (int nt = 0; nt < 8; ++nt)
#pragma unroll
    for (int r = 0; r < 8; ++r) { float x = acc[nt][r]; s1[r] += x; s2[r] += x * x; }
  float mu[8], rs[8];
#pragma unroll
  for (int r = 0; r < 8; ++r) {
    s1[r] = hsum16(s1[r]); s2[r] = hsum16(s2[r]);
    mu[r] = s1[r] * (1.f / 128.f);
    float var = s2[r] * (1.f / 128.f) - mu[r] * mu[r];
    rs[r] = rsqrtf(var + LN_EPS);
  }
#pragma unroll
  for (int nt = 0; nt < 8; ++nt) {
    float gg = nx_g[nt * 16 + ln], bb = nx_b[nt * 16 + ln];
#pragma unroll
    for (int r = 0; r < 8; ++r) {
      float y = (acc[nt][r] - mu[r]) * rs[r] * gg + bb;
      out[(long long)(base + r + 8 * hi) * DIM + nt * 16 + ln] = fmaxf(y, 0.f);
    }
  }
}

// ---------------------------------------------------------------------------
// Launch
// ---------------------------------------------------------------------------
extern "C" void kernel_launch(void* const* d_in, const int* in_sizes, int n_in,
                              void* d_out, int out_size, void* d_ws, size_t ws_size,
                              hipStream_t stream) {
  const float* q    = (const float*)d_in[0];
  const float* mem  = (const float*)d_in[1];
  const float* fk_w = (const float*)d_in[2];
  const float* fk_b = (const float*)d_in[3];
  const float* fv_w = (const float*)d_in[4];
  const float* fv_b = (const float*)d_in[5];
  const float* fx_w = (const float*)d_in[6];
  const float* fx_b = (const float*)d_in[7];
  const float* nk_g = (const float*)d_in[8];
  const float* nk_b = (const float*)d_in[9];
  const float* nv_g = (const float*)d_in[10];
  const float* nv_b = (const float*)d_in[11];
  const float* nx_g = (const float*)d_in[12];
  const float* nx_b = (const float*)d_in[13];

  char* ws = (char*)d_ws;
  const size_t kv_elems   = (size_t)HEADS * MEMN * DIM;                  // 64M
  const size_t part_elems = (size_t)HEADS * NCHUNK * BATCH;              // 128K
  __bf16* k_bf  = (__bf16*)ws;                                           // 128 MiB
  __bf16* v_bf  = (__bf16*)(ws + kv_elems * 2);                          // 128 MiB
  float* o_part = (float*)(ws + kv_elems * 4);                           // 64 MiB
  float* m_part = o_part + part_elems * DIM;
  float* l_part = m_part + part_elems;
  __bf16* x_bf  = (__bf16*)(l_part + part_elems);                        // 512 KiB

  kv_kernel<<<dim3(1024), dim3(256), 0, stream>>>(
      mem, fk_w, fk_b, fv_w, fv_b, nk_g, nk_b, nv_g, nv_b, k_bf, v_bf);
  attn_kernel<<<dim3(NCHUNK, HEADS), dim3(512), 0, stream>>>(
      q, k_bf, v_bf, o_part, m_part, l_part);
  reduce_kernel<<<dim3(HEADS * BATCH), dim3(128), 0, stream>>>(
      o_part, m_part, l_part, x_bf);
  proj_kernel<<<dim3(16), dim3(32), 0, stream>>>(
      x_bf, fx_w, fx_b, nx_g, nx_b, (float*)d_out);
}
